// MPNN3D_13726715478160
// MI455X (gfx1250) — matrix operand order repaired and checked by reading, not compiled, not run on hardware
//
#include <hip/hip_runtime.h>
#include <hip/hip_bf16.h>

// ---------------------------------------------------------------------------
// MPNN on 24^3 grid, HID=128, 6 layers.  Core GEMMs via v_wmma_f32_16x16x32_f16.
// ---------------------------------------------------------------------------

typedef __attribute__((ext_vector_type(16))) _Float16 v16h;
typedef __attribute__((ext_vector_type(8)))  _Float16 h8;
typedef __attribute__((ext_vector_type(8)))  float    v8f;

#define HID   128
#define KPAD  288      // 265 / 261 padded up to multiple of 32
#define DT    0.1f
#define EPSN  1e-5f

__device__ __forceinline__ float swishf(float x) {
    return x / (1.0f + __expf(-x));
}

// A-fragment (16x32 f16) load: lane row = lane%16, half = lane/16.
// elems 0..7  -> K = kb + 8*half + e
// elems 8..15 -> K = kb + 16 + 8*half + (e-8)
__device__ __forceinline__ v16h load_fragA(const _Float16* rowbase, int kb, int half) {
    const _Float16* p = rowbase + kb + 8 * half;
    h8 lo = *(const h8*)(p);
    h8 hi = *(const h8*)(p + 16);
    return __builtin_shufflevector(lo, hi, 0,1,2,3,4,5,6,7,8,9,10,11,12,13,14,15);
}

// B-fragment (32x16 f16) from pre-packed weights: one contiguous 32B per lane.
__device__ __forceinline__ v16h load_fragB(const _Float16* wp, int kbIdx, int nb,
                                           int NT, int lane) {
    return *(const v16h*)(wp + (((long)(kbIdx * NT + nb)) * 32 + lane) * 16);
}

__device__ __forceinline__ v8f wmma_f16(v16h a, v16h b, v8f c) {
    return __builtin_amdgcn_wmma_f32_16x16x32_f16(false, a, false, b, (short)0, c,
                                                  false, false);
}

// ---------------------------------------------------------------------------
// Weight packing: fp32 [L, OUT, Kreal] (torch layout) -> f16 B-fragment order.
// dst element i = layer*KP*OUT + fragId*512 + lane*16 + e
//   fragId = kbIdx*NT + nb ;  K = kbIdx*32 + (e&7) + 16*(e/8) + 8*(lane/16) ; N = nb*16 + lane%16
// ---------------------------------------------------------------------------
__global__ void pack_w_kernel(const float* __restrict__ w, _Float16* __restrict__ dst,
                              int Kreal, int KP, int OUT, int L) {
    long total = (long)L * KP * OUT;
    long i = (long)blockIdx.x * 256 + threadIdx.x;
    if (i >= total) return;
    int perL  = KP * OUT;
    int layer = (int)(i / perL);
    int j     = (int)(i % perL);
    int fragId = j >> 9;
    int lane   = (j >> 4) & 31;
    int e      = j & 15;
    int NT     = OUT >> 4;
    int kbIdx  = fragId / NT;
    int nb     = fragId - kbIdx * NT;
    int k = kbIdx * 32 + (e & 7) + ((e >> 3) << 4) + ((lane >> 4) << 3);
    int n = nb * 16 + (lane & 15);
    float v = (k < Kreal) ? w[(long)layer * OUT * Kreal + (long)n * Kreal + k] : 0.0f;
    dst[i] = (_Float16)v;
}

__global__ void zero_kernel(float* __restrict__ p, int n) {
    int i = blockIdx.x * 256 + threadIdx.x;
    if (i < n) p[i] = 0.0f;
}

// per-graph (bs=1) min/max of grid coords
__global__ void minmax_kernel(const float* __restrict__ g, float* __restrict__ mnmx,
                              int N) {
    __shared__ float smn[3 * 256];
    __shared__ float smx[3 * 256];
    int t = threadIdx.x;
    float mn[3] = {1e30f, 1e30f, 1e30f};
    float mx[3] = {-1e30f, -1e30f, -1e30f};
    for (int n = t; n < N; n += 256) {
        for (int d = 0; d < 3; ++d) {
            float v = g[n * 3 + d];
            mn[d] = fminf(mn[d], v);
            mx[d] = fmaxf(mx[d], v);
        }
    }
    for (int d = 0; d < 3; ++d) { smn[d * 256 + t] = mn[d]; smx[d * 256 + t] = mx[d]; }
    __syncthreads();
    for (int s = 128; s > 0; s >>= 1) {
        if (t < s) {
            for (int d = 0; d < 3; ++d) {
                smn[d * 256 + t] = fminf(smn[d * 256 + t], smn[d * 256 + t + s]);
                smx[d * 256 + t] = fmaxf(smx[d * 256 + t], smx[d * 256 + t + s]);
            }
        }
        __syncthreads();
    }
    if (t < 3) { mnmx[t] = smn[t * 256]; mnmx[3 + t] = smx[t * 256]; }
}

// encoder first layer (K=9, scalar) + pos normalization
__global__ void encoder_kernel(const float* __restrict__ u, const float* __restrict__ g,
                               const float* __restrict__ pars,
                               const float* __restrict__ mnmx,
                               const float* __restrict__ w, const float* __restrict__ b,
                               float* __restrict__ pos, _Float16* __restrict__ x1,
                               int N) {
    int n = blockIdx.x * 256 + threadIdx.x;
    if (n >= N) return;
    float in9[9];
    in9[0] = u[n];
    for (int d = 0; d < 3; ++d) {
        float p = (g[n * 3 + d] - mnmx[d]) / (mnmx[3 + d] - mnmx[d]);
        pos[n * 3 + d] = p;
        in9[1 + d] = p;
    }
    for (int q = 0; q < 5; ++q) in9[4 + q] = pars[n * 5 + q];
    for (int o = 0; o < HID; ++o) {
        float s = b[o];
#pragma unroll
        for (int k = 0; k < 9; ++k) s += w[o * 9 + k] * in9[k];
        x1[(long)n * HID + o] = (_Float16)swishf(s);
    }
}

// Generic WMMA node-GEMM:  out[N, NT*16] = swish(act[N,128] @ W + bias)
// MODE 0 -> f32 out, MODE 1 -> f16 out.  Block = 128 thr (4 waves), 64 rows.
// NT/MODE are compile-time so acc[] stays fully register-resident (no scratch).
template <int NT, int MODE>
__global__ __launch_bounds__(128) void node_gemm_kernel(
    const _Float16* __restrict__ act, const _Float16* __restrict__ wp,
    const float* __restrict__ bias, float* __restrict__ outF,
    _Float16* __restrict__ outH) {
    const int tid  = threadIdx.x;
    const int lane = tid & 31;
    const int w    = tid >> 5;
    const int half = lane >> 4;
    const int lrow = lane & 15;
    const int n0   = blockIdx.x * 64;

    const _Float16* rb = act + (long)(n0 + w * 16 + lrow) * HID;

    v8f vz = {0.f, 0.f, 0.f, 0.f, 0.f, 0.f, 0.f, 0.f};
    v8f acc[NT];
#pragma unroll
    for (int nb = 0; nb < NT; ++nb) acc[nb] = vz;

#pragma unroll
    for (int kbIdx = 0; kbIdx < 4; ++kbIdx) {
        v16h a = load_fragA(rb, kbIdx * 32, half);
#pragma unroll
        for (int nb = 0; nb < NT; ++nb) {
            v16h b = load_fragB(wp, kbIdx, nb, NT, lane);
            acc[nb] = wmma_f16(a, b, acc[nb]);
        }
    }
    const int OUT = NT * 16;
#pragma unroll
    for (int nb = 0; nb < NT; ++nb) {
        int col = nb * 16 + lrow;
        float bs = bias[col];
#pragma unroll
        for (int r = 0; r < 8; ++r) {
            int n = n0 + w * 16 + r + 8 * half;
            float v = swishf(acc[nb][r] + bs);
            if (MODE == 0) outF[(long)n * OUT + col] = v;
            else           outH[(long)n * OUT + col] = (_Float16)v;
        }
    }
}

__global__ void deg_kernel(const int* __restrict__ dstI, float* __restrict__ deg,
                           int E) {
    int i = blockIdx.x * 256 + threadIdx.x;
    if (i < E) atomicAdd(&deg[dstI[i]], 1.0f);
}

__global__ void invdeg_kernel(float* __restrict__ deg, int N) {
    int n = blockIdx.x * 256 + threadIdx.x;
    if (n < N) deg[n] = 1.0f / fmaxf(deg[n], 1.0f);
}

// ---------------------------------------------------------------------------
// Fused message MLP:  per 64-edge block, stage concat rows to LDS (f16),
// GEMM1 (288x128) -> swish -> LDS -> GEMM2 (128x128) -> swish -> atomic scatter.
// ---------------------------------------------------------------------------
__global__ __launch_bounds__(128) void msg_kernel(
    const float* __restrict__ f, const float* __restrict__ u,
    const float* __restrict__ pos, const float* __restrict__ pars,
    const int* __restrict__ srcI, const int* __restrict__ dstI,
    const _Float16* __restrict__ w1p, const float* __restrict__ b1,
    const _Float16* __restrict__ w2p, const float* __restrict__ b2,
    float* __restrict__ agg, int E) {
    __shared__ _Float16 sA[64 * KPAD];
    __shared__ _Float16 sM[64 * HID];

    const int tid = threadIdx.x;
    const int e0  = blockIdx.x * 64;

    // stage A rows: [f_dst(128) | f_src(128) | du(1) | dp(3) | var(5) | pad]
    for (int idx = tid; idx < 64 * KPAD; idx += 128) {
        int r = idx / KPAD;
        int k = idx - r * KPAD;
        int e = e0 + r;
        float v = 0.0f;
        if (e < E) {
            int di = dstI[e], si = srcI[e];
            if (k < 128)        v = f[(long)di * HID + k];
            else if (k < 256)   v = f[(long)si * HID + (k - 128)];
            else if (k == 256)  v = u[di] - u[si];
            else if (k < 260)   v = pos[di * 3 + (k - 257)] - pos[si * 3 + (k - 257)];
            else if (k < 265)   v = pars[di * 5 + (k - 260)];
        }
        sA[idx] = (_Float16)v;
    }
    __syncthreads();

    const int lane = tid & 31;
    const int w    = tid >> 5;
    const int half = lane >> 4;
    const int lrow = lane & 15;
    const int row  = w * 16 + lrow;   // local A row for this lane

    v8f vz = {0.f, 0.f, 0.f, 0.f, 0.f, 0.f, 0.f, 0.f};
    v8f acc[8];
#pragma unroll
    for (int nb = 0; nb < 8; ++nb) acc[nb] = vz;

    const _Float16* rb = &sA[row * KPAD];
#pragma unroll
    for (int kbIdx = 0; kbIdx < KPAD / 32; ++kbIdx) {
        v16h a = load_fragA(rb, kbIdx * 32, half);
        if (kbIdx + 1 < KPAD / 32)
            __builtin_prefetch(w1p + (((long)(kbIdx + 1) * 8) * 32 + lane) * 16, 0, 0);
#pragma unroll
        for (int nb = 0; nb < 8; ++nb) {
            v16h b = load_fragB(w1p, kbIdx, nb, 8, lane);
            acc[nb] = wmma_f16(a, b, acc[nb]);
        }
    }
    // epilogue 1 -> LDS (wave-private rows)
#pragma unroll
    for (int nb = 0; nb < 8; ++nb) {
        int col = nb * 16 + lrow;
        float bs = b1[col];
#pragma unroll
        for (int r = 0; r < 8; ++r) {
            int rr = w * 16 + r + 8 * half;
            sM[rr * HID + col] = (_Float16)swishf(acc[nb][r] + bs);
        }
    }

#pragma unroll
    for (int nb = 0; nb < 8; ++nb) acc[nb] = vz;
    const _Float16* rb2 = &sM[row * HID];
#pragma unroll
    for (int kbIdx = 0; kbIdx < 4; ++kbIdx) {
        v16h a = load_fragA(rb2, kbIdx * 32, half);
#pragma unroll
        for (int nb = 0; nb < 8; ++nb) {
            v16h b = load_fragB(w2p, kbIdx, nb, 8, lane);
            acc[nb] = wmma_f16(a, b, acc[nb]);
        }
    }
    // epilogue 2: swish + mean-aggregation scatter (deg division done at use site)
    int dcache[8];
#pragma unroll
    for (int r = 0; r < 8; ++r) {
        int e = e0 + w * 16 + r + 8 * half;
        dcache[r] = (e < E) ? dstI[e] : -1;
    }
#pragma unroll
    for (int nb = 0; nb < 8; ++nb) {
        int col = nb * 16 + lrow;
        float bs = b2[col];
#pragma unroll
        for (int r = 0; r < 8; ++r) {
            if (dcache[r] >= 0) {
                float v = swishf(acc[nb][r] + bs);
                atomicAdd(&agg[(long)dcache[r] * HID + col], v);
            }
        }
    }
}

// ---------------------------------------------------------------------------
// Fused update MLP + residual + InstanceNorm statistics.
// ---------------------------------------------------------------------------
__global__ __launch_bounds__(128) void upd_kernel(
    float* __restrict__ f, const float* __restrict__ agg,
    const float* __restrict__ invdeg, const float* __restrict__ pars,
    const _Float16* __restrict__ w1p, const float* __restrict__ b1,
    const _Float16* __restrict__ w2p, const float* __restrict__ b2,
    float* __restrict__ chSum, float* __restrict__ chSumSq, int N) {
    __shared__ _Float16 sA[64 * KPAD];
    __shared__ _Float16 sM[64 * HID];

    const int tid = threadIdx.x;
    const int n0  = blockIdx.x * 64;

    // stage: [f(128) | agg/deg(128) | pars(5) | pad]
    for (int idx = tid; idx < 64 * KPAD; idx += 128) {
        int r = idx / KPAD;
        int k = idx - r * KPAD;
        int n = n0 + r;
        float v = 0.0f;
        if (n < N) {
            if (k < 128)       v = f[(long)n * HID + k];
            else if (k < 256)  v = agg[(long)n * HID + (k - 128)] * invdeg[n];
            else if (k < 261)  v = pars[n * 5 + (k - 256)];
        }
        sA[idx] = (_Float16)v;
    }
    __syncthreads();

    const int lane = tid & 31;
    const int w    = tid >> 5;
    const int half = lane >> 4;
    const int lrow = lane & 15;
    const int row  = w * 16 + lrow;

    v8f vz = {0.f, 0.f, 0.f, 0.f, 0.f, 0.f, 0.f, 0.f};
    v8f acc[8];
#pragma unroll
    for (int nb = 0; nb < 8; ++nb) acc[nb] = vz;

    const _Float16* rb = &sA[row * KPAD];
#pragma unroll
    for (int kbIdx = 0; kbIdx < KPAD / 32; ++kbIdx) {
        v16h a = load_fragA(rb, kbIdx * 32, half);
#pragma unroll
        for (int nb = 0; nb < 8; ++nb) {
            v16h b = load_fragB(w1p, kbIdx, nb, 8, lane);
            acc[nb] = wmma_f16(a, b, acc[nb]);
        }
    }
#pragma unroll
    for (int nb = 0; nb < 8; ++nb) {
        int col = nb * 16 + lrow;
        float bs = b1[col];
#pragma unroll
        for (int r = 0; r < 8; ++r) {
            int rr = w * 16 + r + 8 * half;
            sM[rr * HID + col] = (_Float16)swishf(acc[nb][r] + bs);
        }
    }

#pragma unroll
    for (int nb = 0; nb < 8; ++nb) acc[nb] = vz;
    const _Float16* rb2 = &sM[row * HID];
#pragma unroll
    for (int kbIdx = 0; kbIdx < 4; ++kbIdx) {
        v16h a = load_fragA(rb2, kbIdx * 32, half);
#pragma unroll
        for (int nb = 0; nb < 8; ++nb) {
            v16h b = load_fragB(w2p, kbIdx, nb, 8, lane);
            acc[nb] = wmma_f16(a, b, acc[nb]);
        }
    }
    // residual add in place + per-channel moment accumulation
#pragma unroll
    for (int nb = 0; nb < 8; ++nb) {
        int col = nb * 16 + lrow;
        float bs = b2[col];
        float s = 0.0f, sq = 0.0f;
#pragma unroll
        for (int r = 0; r < 8; ++r) {
            int n = n0 + w * 16 + r + 8 * half;
            if (n < N) {
                float fn = f[(long)n * HID + col] + swishf(acc[nb][r] + bs);
                f[(long)n * HID + col] = fn;
                s += fn;
                sq += fn * fn;
            }
        }
        atomicAdd(&chSum[col], s);
        atomicAdd(&chSumSq[col], sq);
    }
}

__global__ void norm_kernel(float* __restrict__ f, const float* __restrict__ chSum,
                            const float* __restrict__ chSumSq, int N) {
    long i = (long)blockIdx.x * 256 + threadIdx.x;
    if (i >= (long)N * HID) return;
    int c = (int)(i & (HID - 1));
    float inv_n = 1.0f / (float)N;
    float mu = chSum[c] * inv_n;
    float var = chSumSq[c] * inv_n - mu * mu;
    f[i] = (f[i] - mu) * rsqrtf(var + EPSN);
}

__global__ void f2h_kernel(const float* __restrict__ in, _Float16* __restrict__ out,
                           int n) {
    long i = (long)blockIdx.x * 256 + threadIdx.x;
    if (i < n) out[i] = (_Float16)in[i];
}

__global__ void decoder_kernel(const _Float16* __restrict__ xd,
                               const float* __restrict__ w, const float* __restrict__ b,
                               const float* __restrict__ u, float* __restrict__ out,
                               int N) {
    int n = blockIdx.x * 256 + threadIdx.x;
    if (n >= N) return;
    float s = b[0];
#pragma unroll
    for (int k = 0; k < 64; ++k) s += (float)xd[(long)n * 64 + k] * w[k];
    out[n] = u[n] + DT * s;
}

// ---------------------------------------------------------------------------
extern "C" void kernel_launch(void* const* d_in, const int* in_sizes, int n_in,
                              void* d_out, int out_size, void* d_ws, size_t ws_size,
                              hipStream_t stream) {
    if (n_in < 21) return;
    const float* uIn   = (const float*)d_in[0];
    const float* pars  = (const float*)d_in[1];
    // d_in[2] mask: all-ones, unused
    const float* grid  = (const float*)d_in[3];
    const int*   ei    = (const int*)d_in[4];
    const float* emb1w = (const float*)d_in[5];
    const float* emb1b = (const float*)d_in[6];
    const float* emb2w = (const float*)d_in[7];
    const float* emb2b = (const float*)d_in[8];
    const float* out1w = (const float*)d_in[9];
    const float* out1b = (const float*)d_in[10];
    const float* out2w = (const float*)d_in[11];
    const float* out2b = (const float*)d_in[12];
    const float* msg1w = (const float*)d_in[13];
    const float* msg1b = (const float*)d_in[14];
    const float* msg2w = (const float*)d_in[15];
    const float* msg2b = (const float*)d_in[16];
    const float* upd1w = (const float*)d_in[17];
    const float* upd1b = (const float*)d_in[18];
    const float* upd2w = (const float*)d_in[19];
    const float* upd2b = (const float*)d_in[20];

    const int N = in_sizes[0];
    const int E = in_sizes[4] / 2;
    const int* srcI = ei;
    const int* dstI = ei + E;

    // workspace carve-out
    char* wsp = (char*)d_ws;
    auto carve = [&](size_t bytes) -> void* {
        void* p = (void*)wsp;
        wsp += (bytes + 255) & ~(size_t)255;
        return p;
    };
    float*     pos   = (float*)carve((size_t)N * 3 * 4);
    float*     deg   = (float*)carve((size_t)N * 4);
    float*     mnmx  = (float*)carve(256);
    float*     chS   = (float*)carve(512 * 4);          // chSum[128] | chSumSq[128]
    float*     chQ   = chS + 128;
    _Float16*  x1    = (_Float16*)carve((size_t)N * HID * 2);
    _Float16*  xdec  = (_Float16*)carve((size_t)N * 64 * 2);
    float*     f     = (float*)carve((size_t)N * HID * 4);
    float*     agg   = (float*)carve((size_t)N * HID * 4);
    _Float16*  emb2p = (_Float16*)carve((size_t)128 * 128 * 2);
    _Float16*  out1p = (_Float16*)carve((size_t)128 * 64 * 2);
    _Float16*  m1p   = (_Float16*)carve((size_t)6 * KPAD * 128 * 2);
    _Float16*  m2p   = (_Float16*)carve((size_t)6 * 128 * 128 * 2);
    _Float16*  u1p   = (_Float16*)carve((size_t)6 * KPAD * 128 * 2);
    _Float16*  u2p   = (_Float16*)carve((size_t)6 * 128 * 128 * 2);

    auto blocks = [](long n, int b) { return (unsigned)((n + b - 1) / b); };

    // --- weight packing (fp32 -> f16 fragment order) ---
    pack_w_kernel<<<blocks((long)128 * 128, 256), 256, 0, stream>>>(emb2w, emb2p, 128, 128, 128, 1);
    pack_w_kernel<<<blocks((long)128 * 64, 256), 256, 0, stream>>>(out1w, out1p, 128, 128, 64, 1);
    pack_w_kernel<<<blocks((long)6 * KPAD * 128, 256), 256, 0, stream>>>(msg1w, m1p, 265, KPAD, 128, 6);
    pack_w_kernel<<<blocks((long)6 * 128 * 128, 256), 256, 0, stream>>>(msg2w, m2p, 128, 128, 128, 6);
    pack_w_kernel<<<blocks((long)6 * KPAD * 128, 256), 256, 0, stream>>>(upd1w, u1p, 261, KPAD, 128, 6);
    pack_w_kernel<<<blocks((long)6 * 128 * 128, 256), 256, 0, stream>>>(upd2w, u2p, 128, 128, 128, 6);

    // --- encoder ---
    minmax_kernel<<<1, 256, 0, stream>>>(grid, mnmx, N);
    encoder_kernel<<<blocks(N, 256), 256, 0, stream>>>(uIn, grid, pars, mnmx, emb1w,
                                                       emb1b, pos, x1, N);
    node_gemm_kernel<8, 0><<<blocks(N, 64), 128, 0, stream>>>(x1, emb2p, emb2b, f,
                                                              nullptr);

    // --- degree / inverse degree ---
    zero_kernel<<<blocks(N, 256), 256, 0, stream>>>(deg, N);
    deg_kernel<<<blocks(E, 256), 256, 0, stream>>>(dstI, deg, E);
    invdeg_kernel<<<blocks(N, 256), 256, 0, stream>>>(deg, N);

    // --- 6 GNN layers ---
    for (int l = 0; l < 6; ++l) {
        zero_kernel<<<blocks((long)N * HID, 256), 256, 0, stream>>>(agg, N * HID);
        zero_kernel<<<1, 256, 0, stream>>>(chS, 256);
        msg_kernel<<<blocks(E, 64), 128, 0, stream>>>(
            f, uIn, pos, pars, srcI, dstI,
            m1p + (long)l * KPAD * 128, msg1b + l * 128,
            m2p + (long)l * 128 * 128, msg2b + l * 128, agg, E);
        upd_kernel<<<blocks(N, 64), 128, 0, stream>>>(
            f, agg, deg, pars,
            u1p + (long)l * KPAD * 128, upd1b + l * 128,
            u2p + (long)l * 128 * 128, upd2b + l * 128, chS, chQ, N);
        norm_kernel<<<blocks((long)N * HID, 256), 256, 0, stream>>>(f, chS, chQ, N);
    }

    // --- decoder ---
    f2h_kernel<<<blocks((long)N * HID, 256), 256, 0, stream>>>(f, x1, N * HID);
    node_gemm_kernel<4, 1><<<blocks(N, 64), 128, 0, stream>>>(x1, out1p, out1b,
                                                              nullptr, xdec);
    decoder_kernel<<<blocks(N, 256), 256, 0, stream>>>(xdec, out2w, out2b, uIn,
                                                       (float*)d_out, N);
}
